// ClassifierGNN_34007551050153
// MI455X (gfx1250) — matrix operand order repaired and checked by reading, not compiled, not run on hardware
//
#include <hip/hip_runtime.h>
#include <stdint.h>

// ---------------------------------------------------------------------------
// GVP-GNN for MI455X (gfx1250).
//  * Edge-message GVP stack fused into one kernel per layer: 32-edge tiles
//    (2 WMMA M-tiles -> each B fragment feeds 2 v_wmma ops, halving L2
//    weight re-reads), gather->LDS bf16 A-strip, three
//    v_wmma_f32_16x16x32_bf16 GEMM stages, gated vector channels in VALU,
//    scatter via global_atomic_add_f32.
//  * Node scalar state kept BOTH fp32 (residual/LN) and bf16 (GEMM gather):
//    halves the dominant gather traffic with no extra quantization error.
//  * Node feed-forward GVPs as a second fused WMMA kernel (16-node tiles).
//  * Edge in-degrees computed once (shared by all 3 layers).
//  * wave32 everywhere: 256-thread blocks = 8 waves; shuffle reductions.
// ---------------------------------------------------------------------------

#define NNODES  30000
#define NEDGES  900000
#define NGRAPHS 64
#define NSD     100     // node scalar dim
#define NSP     112     // padded bf16 row (224B, 16B aligned)
#define NVD     16      // node vector dim

typedef unsigned short ushort_t;
typedef __attribute__((ext_vector_type(16))) __bf16 v16bf;
typedef __attribute__((ext_vector_type(8)))  float  v8f;

union FragU { uint4 u[2]; v16bf v; };

__device__ __forceinline__ ushort_t f2bf(float x) {
  union { float f; unsigned int u; } c; c.f = x;
  unsigned int r = c.u + 0x7FFFu + ((c.u >> 16) & 1u);   // round-to-nearest-even
  return (ushort_t)(r >> 16);
}
__device__ __forceinline__ float sigmoidf_(float x) { return 1.f / (1.f + __expf(-x)); }

// A fragment (16x32 bf16, M rows): lane L -> M = L&15; lane half selects K
// groups {0..7,16..23} vs {8..15,24..31}.  Source is row-major [16][kp].
__device__ __forceinline__ v16bf loadFragA(const ushort_t* a, int kp, int lane) {
  int m = lane & 15, kh = (lane >> 4) << 3;
  FragU f;
  const ushort_t* p = a + m * kp + kh;
  f.u[0] = *(const uint4*)(p);
  f.u[1] = *(const uint4*)(p + 16);
  return f.v;
}
// B fragment (32x16 bf16, K x N): lane L -> N = L&15; lane half selects K
// groups {0..7,16..23} vs {8..15,24..31}.  Source is transposed weights Bt[N][kp] row-major.
__device__ __forceinline__ v16bf loadFragB(const ushort_t* b, int kp, int lane) {
  int n = lane & 15, kh = (lane >> 4) << 3;
  FragU f;
  const ushort_t* p = b + n * kp + kh;
  f.u[0] = *(const uint4*)(p);
  f.u[1] = *(const uint4*)(p + 16);
  return f.v;
}
__device__ __forceinline__ v8f wmma_bf16(v16bf a, v16bf b, v8f c) {
  return __builtin_amdgcn_wmma_f32_16x16x32_bf16(false, a, false, b, (short)0, c,
                                                 false, false);
}

// ---------------------------------------------------------------------------
// Weight repack: src fp32 [K][N] -> dst bf16 [Np][Kp] (transposed, zero pad)
// ---------------------------------------------------------------------------
__global__ __launch_bounds__(256) void pack_wT(const float* __restrict__ src,
                                               ushort_t* __restrict__ dst,
                                               int K, int N, int Kp, int Np) {
  int idx = blockIdx.x * 256 + threadIdx.x;
  if (idx >= Np * Kp) return;
  int n = idx / Kp, k = idx % Kp;
  float v = (n < N && k < K) ? src[k * N + n] : 0.f;
  dst[idx] = f2bf(v);
}

// in-degree of every node (shared by all layers)
__global__ __launch_bounds__(256) void count_edges(const int* __restrict__ dst,
                                                   float* __restrict__ cnt) {
  int e = blockIdx.x * 256 + threadIdx.x;
  if (e < NEDGES) atomicAdd(&cnt[dst[e]], 1.0f);
}

// ---------------------------------------------------------------------------
// Node embedding: LN(6) + GVP(6,3 -> 100,16), no activations
// ---------------------------------------------------------------------------
__global__ __launch_bounds__(256) void node_embed(
    const float* __restrict__ ns, const float* __restrict__ nv,
    const float* __restrict__ gamma, const float* __restrict__ beta,
    const float* __restrict__ wh, const float* __restrict__ wsM,
    const float* __restrict__ bs, const float* __restrict__ wv,
    float* __restrict__ s_out, ushort_t* __restrict__ s_bf,
    float* __restrict__ v_out) {
  int n = blockIdx.x * 256 + threadIdx.x;
  if (n >= NNODES) return;
  float x[6], mu = 0.f;
  for (int i = 0; i < 6; i++) { x[i] = ns[n * 6 + i]; mu += x[i]; }
  mu *= (1.f / 6.f);
  float var = 0.f;
  for (int i = 0; i < 6; i++) { float d = x[i] - mu; var += d * d; }
  var *= (1.f / 6.f);
  float rs = rsqrtf(var + 1e-5f);
  float scat[22];
  for (int i = 0; i < 6; i++) scat[i] = (x[i] - mu) * rs * gamma[i] + beta[i];
  float v[3][3]; float vss = 0.f;
  for (int i = 0; i < 3; i++) {
    float sq = 0.f;
    for (int c = 0; c < 3; c++) { v[i][c] = nv[n * 9 + i * 3 + c]; sq += v[i][c] * v[i][c]; }
    vss += fmaxf(sq, 1e-8f);
  }
  float rvn = rsqrtf(vss * (1.f / 3.f));
  for (int i = 0; i < 3; i++) for (int c = 0; c < 3; c++) v[i][c] *= rvn;
  float vh[16][3];
  for (int h = 0; h < 16; h++) {
    float sq = 0.f;
    for (int c = 0; c < 3; c++) {
      float a = 0.f;
      for (int i = 0; i < 3; i++) a += v[i][c] * wh[i * 16 + h];
      vh[h][c] = a; sq += a * a;
    }
    scat[6 + h] = sqrtf(fmaxf(sq, 1e-8f));
  }
  for (int c = 0; c < NSD; c++) {
    float a = bs[c];
    for (int k = 0; k < 22; k++) a += scat[k] * wsM[k * NSD + c];
    s_out[(size_t)n * NSD + c] = a;
    s_bf[(size_t)n * NSP + c] = f2bf(a);
  }
  for (int c = NSD; c < NSP; c++) s_bf[(size_t)n * NSP + c] = 0;   // pad
  for (int o = 0; o < NVD; o++)
    for (int c = 0; c < 3; c++) {
      float a = 0.f;
      for (int h = 0; h < 16; h++) a += vh[h][c] * wv[h * 16 + o];
      v_out[(size_t)n * 48 + o * 3 + c] = a;
    }
}

// ---------------------------------------------------------------------------
// Edge embedding: LN(32) + GVP(32,1 -> 32,1); es stored as bf16 (GEMM input)
// ---------------------------------------------------------------------------
__global__ __launch_bounds__(256) void edge_embed(
    const float* __restrict__ esrc, const float* __restrict__ evsrc,
    const float* __restrict__ gamma, const float* __restrict__ beta,
    const float* __restrict__ wh, const float* __restrict__ wsM,
    const float* __restrict__ bs, const float* __restrict__ wv,
    ushort_t* __restrict__ es_bf, float* __restrict__ ev_out) {
  int e = blockIdx.x * 256 + threadIdx.x;
  if (e >= NEDGES) return;
  float x[32], mu = 0.f;
  for (int i = 0; i < 32; i++) { x[i] = esrc[(size_t)e * 32 + i]; mu += x[i]; }
  mu *= (1.f / 32.f);
  float var = 0.f;
  for (int i = 0; i < 32; i++) { float d = x[i] - mu; var += d * d; }
  var *= (1.f / 32.f);
  float rs = rsqrtf(var + 1e-5f);
  float scat[33];
  for (int i = 0; i < 32; i++) scat[i] = (x[i] - mu) * rs * gamma[i] + beta[i];
  float v[3], sq = 0.f;
  for (int c = 0; c < 3; c++) { v[c] = evsrc[(size_t)e * 3 + c]; sq += v[c] * v[c]; }
  float rvn = rsqrtf(fmaxf(sq, 1e-8f));
  float vh[3], hsq = 0.f;
  for (int c = 0; c < 3; c++) { vh[c] = v[c] * rvn * wh[0]; hsq += vh[c] * vh[c]; }
  scat[32] = sqrtf(fmaxf(hsq, 1e-8f));
  for (int c = 0; c < 32; c++) {
    float a = bs[c];
    for (int k = 0; k < 33; k++) a += scat[k] * wsM[k * 32 + c];
    es_bf[(size_t)e * 32 + c] = f2bf(a);
  }
  for (int c = 0; c < 3; c++) ev_out[(size_t)e * 3 + c] = vh[c] * wv[0];
}

// ---------------------------------------------------------------------------
// Fused edge-message kernel: one block = 32 edges (two 16-row WMMA M-tiles),
// 8 waves.  GVP0(265,33 -> 100,16, relu/sig) -> GVP1(116,16 -> 100,16,
// relu/sig) -> GVP2(116,16 -> 100,16) -> atomic scatter into agg_s/agg_v.
// ---------------------------------------------------------------------------
__global__ __launch_bounds__(256) void edge_msg(
    const ushort_t* __restrict__ s_bf, const float* __restrict__ v_node,
    const ushort_t* __restrict__ es_bf, const float* __restrict__ ev,
    const int* __restrict__ src, const int* __restrict__ dst,
    const float* __restrict__ wh0, const float* __restrict__ bs0, const float* __restrict__ wv0,
    const float* __restrict__ wh1, const float* __restrict__ bs1, const float* __restrict__ wv1,
    const float* __restrict__ wh2, const float* __restrict__ bs2, const float* __restrict__ wv2,
    const ushort_t* __restrict__ Bt0, const ushort_t* __restrict__ Bt1,
    const ushort_t* __restrict__ Bt2,
    float* __restrict__ agg_s, float* __restrict__ agg_v) {
  __shared__ __align__(16) ushort_t Abuf[32][288];   // GVP0 A-strip (also A3 alias)
  __shared__ __align__(16) ushort_t A2[32][128];     // GVP1 A-strip
  __shared__ float vhb[32][33][3];
  __shared__ float v1b[32][16][3];
  __shared__ int sid[32], did[32];

  const int tid = threadIdx.x, lane = tid & 31, w = tid >> 5;
  const int half = tid >> 4, hl = tid & 15;          // 16 half-waves, 2 edges each
  const int eb = blockIdx.x * 32;
  ushort_t(*A3)[128] = (ushort_t(*)[128]) & Abuf[0][0];

  if (tid < 32) {
    int e = eb + tid;
    sid[tid] = src[e];
    did[tid] = dst[e];
  }
  for (int i = tid; i < 32 * 23; i += 256) Abuf[i / 23][265 + i % 23] = 0;
  for (int i = tid; i < 32 * 12; i += 256) A2[i / 12][116 + i % 12] = 0;
  __syncthreads();

  // gather scalar features as bf16 pairs: [s_src(100) | es(32) | s_dst(100)]
  for (int i = tid; i < 32 * 50; i += 256) {           // s_src
    int r = i / 50, u = (i % 50) * 2;
    *(unsigned int*)&Abuf[r][u] = *(const unsigned int*)&s_bf[(size_t)sid[r] * NSP + u];
  }
  for (int i = tid; i < 32 * 16; i += 256) {           // es
    int r = i / 16, u = (i % 16) * 2;
    *(unsigned int*)&Abuf[r][100 + u] = *(const unsigned int*)&es_bf[(size_t)(eb + r) * 32 + u];
  }
  for (int i = tid; i < 32 * 50; i += 256) {           // s_dst
    int r = i / 50, u = (i % 50) * 2;
    *(unsigned int*)&Abuf[r][132 + u] = *(const unsigned int*)&s_bf[(size_t)did[r] * NSP + u];
  }
  // GVP0 vector path: vh = concat(v_src, ev, v_dst) @ wh0 (33x33); vn -> A 232..264
  for (int ee = 0; ee < 2; ++ee) {
    int el = half * 2 + ee;
    int e = eb + el, si = sid[el], di = did[el];
    for (int hb = 0; hb < 3; ++hb) {
      int h = hb * 16 + hl;
      if (h < 33) {
        float a0 = 0, a1 = 0, a2 = 0;
        for (int i = 0; i < 33; ++i) {
          float ww = wh0[i * 33 + h];
          const float* p;
          if (i < 16)       p = &v_node[(size_t)si * 48 + i * 3];
          else if (i == 16) p = &ev[(size_t)e * 3];
          else              p = &v_node[(size_t)di * 48 + (i - 17) * 3];
          a0 += ww * p[0]; a1 += ww * p[1]; a2 += ww * p[2];
        }
        vhb[el][h][0] = a0; vhb[el][h][1] = a1; vhb[el][h][2] = a2;
        Abuf[el][232 + h] = f2bf(sqrtf(fmaxf(a0 * a0 + a1 * a1 + a2 * a2, 1e-8f)));
      }
    }
  }
  __syncthreads();

  // GEMM1: relu([32x288] @ Bt0 + bs0) -> A2 cols 0..99
  for (int nt = w; nt < 7; nt += 8) {
    __builtin_prefetch(Bt0 + (size_t)nt * 16 * 288, 0, 1);
    v8f acc0 = {0.f, 0.f, 0.f, 0.f, 0.f, 0.f, 0.f, 0.f};
    v8f acc1 = acc0;
    for (int kt = 0; kt < 9; ++kt) {
      v16bf b = loadFragB(Bt0 + (size_t)nt * 16 * 288 + kt * 32, 288, lane);
      v16bf a0 = loadFragA(&Abuf[0][0] + kt * 32, 288, lane);
      v16bf a1 = loadFragA(&Abuf[16][0] + kt * 32, 288, lane);
      acc0 = wmma_bf16(a0, b, acc0);
      acc1 = wmma_bf16(a1, b, acc1);
    }
    int col = nt * 16 + (lane & 15);
    if (col < 100) {
      float bias = bs0[col];
      int rbase = (lane >> 4) * 8;
      for (int r = 0; r < 8; ++r) {
        A2[rbase + r][col]      = f2bf(fmaxf(acc0[r] + bias, 0.f));
        A2[16 + rbase + r][col] = f2bf(fmaxf(acc1[r] + bias, 0.f));
      }
    }
  }
  __syncthreads();

  // GVP0 v_out (sigmoid-gated) -> v1b; zero A3 pad cols (Abuf now free)
  for (int ee = 0; ee < 2; ++ee) {
    int el = half * 2 + ee;
    float a0 = 0, a1 = 0, a2 = 0;
    for (int h = 0; h < 33; ++h) {
      float ww = wv0[h * 16 + hl];
      a0 += ww * vhb[el][h][0]; a1 += ww * vhb[el][h][1]; a2 += ww * vhb[el][h][2];
    }
    float g = sigmoidf_(sqrtf(fmaxf(a0 * a0 + a1 * a1 + a2 * a2, 1e-8f)));
    v1b[el][hl][0] = a0 * g; v1b[el][hl][1] = a1 * g; v1b[el][hl][2] = a2 * g;
  }
  for (int i = tid; i < 32 * 12; i += 256) A3[i / 12][116 + i % 12] = 0;
  __syncthreads();
  // GVP1 vh = v1 @ wh1 (16x16); vn -> A2 cols 100..115
  for (int ee = 0; ee < 2; ++ee) {
    int el = half * 2 + ee;
    float a0 = 0, a1 = 0, a2 = 0;
    for (int i = 0; i < 16; ++i) {
      float ww = wh1[i * 16 + hl];
      a0 += ww * v1b[el][i][0]; a1 += ww * v1b[el][i][1]; a2 += ww * v1b[el][i][2];
    }
    vhb[el][hl][0] = a0; vhb[el][hl][1] = a1; vhb[el][hl][2] = a2;
    A2[el][100 + hl] = f2bf(sqrtf(fmaxf(a0 * a0 + a1 * a1 + a2 * a2, 1e-8f)));
  }
  __syncthreads();

  // GEMM2: relu([32x128] @ Bt1 + bs1) -> A3 cols 0..99
  for (int nt = w; nt < 7; nt += 8) {
    v8f acc0 = {0.f, 0.f, 0.f, 0.f, 0.f, 0.f, 0.f, 0.f};
    v8f acc1 = acc0;
    for (int kt = 0; kt < 4; ++kt) {
      v16bf b = loadFragB(Bt1 + (size_t)nt * 16 * 128 + kt * 32, 128, lane);
      v16bf a0 = loadFragA(&A2[0][0] + kt * 32, 128, lane);
      v16bf a1 = loadFragA(&A2[16][0] + kt * 32, 128, lane);
      acc0 = wmma_bf16(a0, b, acc0);
      acc1 = wmma_bf16(a1, b, acc1);
    }
    int col = nt * 16 + (lane & 15);
    if (col < 100) {
      float bias = bs1[col];
      int rbase = (lane >> 4) * 8;
      for (int r = 0; r < 8; ++r) {
        A3[rbase + r][col]      = f2bf(fmaxf(acc0[r] + bias, 0.f));
        A3[16 + rbase + r][col] = f2bf(fmaxf(acc1[r] + bias, 0.f));
      }
    }
  }
  __syncthreads();

  // GVP1 v_out (gated) -> v1b
  for (int ee = 0; ee < 2; ++ee) {
    int el = half * 2 + ee;
    float a0 = 0, a1 = 0, a2 = 0;
    for (int h = 0; h < 16; ++h) {
      float ww = wv1[h * 16 + hl];
      a0 += ww * vhb[el][h][0]; a1 += ww * vhb[el][h][1]; a2 += ww * vhb[el][h][2];
    }
    float g = sigmoidf_(sqrtf(fmaxf(a0 * a0 + a1 * a1 + a2 * a2, 1e-8f)));
    v1b[el][hl][0] = a0 * g; v1b[el][hl][1] = a1 * g; v1b[el][hl][2] = a2 * g;
  }
  __syncthreads();
  // GVP2 vh; vn -> A3 cols 100..115
  for (int ee = 0; ee < 2; ++ee) {
    int el = half * 2 + ee;
    float a0 = 0, a1 = 0, a2 = 0;
    for (int i = 0; i < 16; ++i) {
      float ww = wh2[i * 16 + hl];
      a0 += ww * v1b[el][i][0]; a1 += ww * v1b[el][i][1]; a2 += ww * v1b[el][i][2];
    }
    vhb[el][hl][0] = a0; vhb[el][hl][1] = a1; vhb[el][hl][2] = a2;
    A3[el][100 + hl] = f2bf(sqrtf(fmaxf(a0 * a0 + a1 * a1 + a2 * a2, 1e-8f)));
  }
  __syncthreads();

  // GEMM3 (no act) + scalar scatter
  for (int nt = w; nt < 7; nt += 8) {
    v8f acc0 = {0.f, 0.f, 0.f, 0.f, 0.f, 0.f, 0.f, 0.f};
    v8f acc1 = acc0;
    for (int kt = 0; kt < 4; ++kt) {
      v16bf b = loadFragB(Bt2 + (size_t)nt * 16 * 128 + kt * 32, 128, lane);
      v16bf a0 = loadFragA(&A3[0][0] + kt * 32, 128, lane);
      v16bf a1 = loadFragA(&A3[16][0] + kt * 32, 128, lane);
      acc0 = wmma_bf16(a0, b, acc0);
      acc1 = wmma_bf16(a1, b, acc1);
    }
    int col = nt * 16 + (lane & 15);
    if (col < 100) {
      float bias = bs2[col];
      int rbase = (lane >> 4) * 8;
      for (int r = 0; r < 8; ++r) {
        atomicAdd(&agg_s[(size_t)did[rbase + r] * NSD + col], acc0[r] + bias);
        atomicAdd(&agg_s[(size_t)did[16 + rbase + r] * NSD + col], acc1[r] + bias);
      }
    }
  }
  // GVP2 v_out (no gating) + vector scatter
  for (int ee = 0; ee < 2; ++ee) {
    int el = half * 2 + ee;
    float a0 = 0, a1 = 0, a2 = 0;
    for (int h = 0; h < 16; ++h) {
      float ww = wv2[h * 16 + hl];
      a0 += ww * vhb[el][h][0]; a1 += ww * vhb[el][h][1]; a2 += ww * vhb[el][h][2];
    }
    size_t bi = (size_t)did[el] * 48 + hl * 3;
    atomicAdd(&agg_v[bi + 0], a0);
    atomicAdd(&agg_v[bi + 1], a1);
    atomicAdd(&agg_v[bi + 2], a2);
  }
}

// ---------------------------------------------------------------------------
// Node update after message aggregation: residual + gvp_layernorm (norm0)
// One wave per node (wave32).  Refreshes both fp32 and bf16 scalar state.
// ---------------------------------------------------------------------------
__global__ __launch_bounds__(256) void node_update(
    float* __restrict__ s_node, ushort_t* __restrict__ s_bf,
    float* __restrict__ v_node,
    const float* __restrict__ agg_s, const float* __restrict__ agg_v,
    const float* __restrict__ cnt, const float* __restrict__ gamma,
    const float* __restrict__ beta) {
  int node = blockIdx.x * 8 + (threadIdx.x >> 5);
  int lane = threadIdx.x & 31;
  if (node >= NNODES) return;
  float inv = 1.f / fmaxf(cnt[node], 1.f);
  float x[4], sum = 0.f, sumsq = 0.f;
  for (int k = 0; k < 4; ++k) {
    int col = lane + 32 * k;
    float v = 0.f;
    if (col < NSD)
      v = s_node[(size_t)node * NSD + col] + agg_s[(size_t)node * NSD + col] * inv;
    x[k] = v; sum += v; sumsq += v * v;
  }
  for (int m = 16; m; m >>= 1) { sum += __shfl_xor(sum, m, 32); sumsq += __shfl_xor(sumsq, m, 32); }
  float mu = sum / NSD, var = sumsq / NSD - mu * mu;
  float rs = rsqrtf(var + 1e-5f);
  for (int k = 0; k < 4; ++k) {
    int col = lane + 32 * k;
    if (col < NSD) {
      float o = (x[k] - mu) * rs * gamma[col] + beta[col];
      s_node[(size_t)node * NSD + col] = o;
      s_bf[(size_t)node * NSP + col] = f2bf(o);
    }
  }
  float vv[3] = {0.f, 0.f, 0.f}, sq = 0.f;
  if (lane < 16) {
    for (int c = 0; c < 3; ++c) {
      vv[c] = v_node[(size_t)node * 48 + lane * 3 + c] +
              agg_v[(size_t)node * 48 + lane * 3 + c] * inv;
      sq += vv[c] * vv[c];
    }
    sq = fmaxf(sq, 1e-8f);
  }
  float tot = sq;
  for (int m = 8; m; m >>= 1) tot += __shfl_xor(tot, m, 32);
  if (lane < 16) {
    float r = rsqrtf(tot * (1.f / 16.f));
    for (int c = 0; c < 3; ++c)
      v_node[(size_t)node * 48 + lane * 3 + c] = vv[c] * r;
  }
}

// ---------------------------------------------------------------------------
// Fused node feed-forward: ff0(132,16 -> 400,32 relu/sig), ff1(432,32 -> 100,16),
// residual + gvp_layernorm (norm1).  One block = 16 nodes.
// ---------------------------------------------------------------------------
__global__ __launch_bounds__(256) void node_ff(
    float* __restrict__ s_node, ushort_t* __restrict__ s_bf,
    float* __restrict__ v_node,
    const float* __restrict__ whf0, const float* __restrict__ bf0, const float* __restrict__ wvf0,
    const float* __restrict__ whf1, const float* __restrict__ bf1, const float* __restrict__ wvf1,
    const ushort_t* __restrict__ Btf0, const ushort_t* __restrict__ Btf1,
    const float* __restrict__ gamma, const float* __restrict__ beta) {
  __shared__ __align__(16) ushort_t Aff[16][160];
  __shared__ __align__(16) ushort_t A2f[16][448];
  __shared__ float vhf[16][32][3];
  __shared__ float v1f[16][32][3];
  __shared__ float sfin[16][112];

  const int tid = threadIdx.x, lane = tid & 31, w = tid >> 5;
  const int nb = blockIdx.x * 16;

  for (int i = tid; i < 16 * 50; i += 256) {
    int r = i / 50, u = (i % 50) * 2;
    *(unsigned int*)&Aff[r][u] = *(const unsigned int*)&s_bf[(size_t)(nb + r) * NSP + u];
  }
  for (int i = tid; i < 16 * 28; i += 256) Aff[i / 28][132 + i % 28] = 0;
  for (int i = tid; i < 16 * 16; i += 256) A2f[i / 16][432 + i % 16] = 0;
  // ff0 vector path: vh = v @ whf0 (16->32); vn -> Aff cols 100..131
  {
    int nl = w * 2 + (lane >> 4);
    int n = nb + nl, hlf = lane & 15;
    for (int hb = 0; hb < 2; ++hb) {
      int h = hb * 16 + hlf;
      float a0 = 0, a1 = 0, a2 = 0;
      for (int i = 0; i < 16; ++i) {
        float ww = whf0[i * 32 + h];
        const float* p = &v_node[(size_t)n * 48 + i * 3];
        a0 += ww * p[0]; a1 += ww * p[1]; a2 += ww * p[2];
      }
      vhf[nl][h][0] = a0; vhf[nl][h][1] = a1; vhf[nl][h][2] = a2;
      Aff[nl][100 + h] = f2bf(sqrtf(fmaxf(a0 * a0 + a1 * a1 + a2 * a2, 1e-8f)));
    }
  }
  __syncthreads();
  // GEMM ff0: relu([16x160] @ Btf0 + bf0) -> A2f cols 0..399
  for (int nt = w; nt < 26; nt += 8) {
    v8f acc = {0.f, 0.f, 0.f, 0.f, 0.f, 0.f, 0.f, 0.f};
    for (int kt = 0; kt < 5; ++kt) {
      v16bf a = loadFragA(&Aff[0][0] + kt * 32, 160, lane);
      v16bf b = loadFragB(Btf0 + (size_t)nt * 16 * 160 + kt * 32, 160, lane);
      acc = wmma_bf16(a, b, acc);
    }
    int col = nt * 16 + (lane & 15);
    if (col < 400) {
      float bias = bf0[col];
      int rbase = (lane >> 4) * 8;
      for (int r = 0; r < 8; ++r)
        A2f[rbase + r][col] = f2bf(fmaxf(acc[r] + bias, 0.f));
    }
  }
  __syncthreads();
  // ff0 v_out gated -> v1f
  {
    int nl = w * 2 + (lane >> 4);
    int ol = lane & 15;
    for (int ob = 0; ob < 2; ++ob) {
      int o = ob * 16 + ol;
      float a0 = 0, a1 = 0, a2 = 0;
      for (int h = 0; h < 32; ++h) {
        float ww = wvf0[h * 32 + o];
        a0 += ww * vhf[nl][h][0]; a1 += ww * vhf[nl][h][1]; a2 += ww * vhf[nl][h][2];
      }
      float g = sigmoidf_(sqrtf(fmaxf(a0 * a0 + a1 * a1 + a2 * a2, 1e-8f)));
      v1f[nl][o][0] = a0 * g; v1f[nl][o][1] = a1 * g; v1f[nl][o][2] = a2 * g;
    }
  }
  __syncthreads();
  // ff1 vh = v1 @ whf1 (32x32); vn -> A2f cols 400..431
  {
    int nl = w * 2 + (lane >> 4);
    int hlf = lane & 15;
    for (int hb = 0; hb < 2; ++hb) {
      int h = hb * 16 + hlf;
      float a0 = 0, a1 = 0, a2 = 0;
      for (int i = 0; i < 32; ++i) {
        float ww = whf1[i * 32 + h];
        a0 += ww * v1f[nl][i][0]; a1 += ww * v1f[nl][i][1]; a2 += ww * v1f[nl][i][2];
      }
      vhf[nl][h][0] = a0; vhf[nl][h][1] = a1; vhf[nl][h][2] = a2;
      A2f[nl][400 + h] = f2bf(sqrtf(fmaxf(a0 * a0 + a1 * a1 + a2 * a2, 1e-8f)));
    }
  }
  __syncthreads();
  // GEMM ff1 (no act): [16x448] @ Btf1 + bf1 -> sfin
  for (int nt = w; nt < 7; nt += 8) {
    v8f acc = {0.f, 0.f, 0.f, 0.f, 0.f, 0.f, 0.f, 0.f};
    for (int kt = 0; kt < 14; ++kt) {
      v16bf a = loadFragA(&A2f[0][0] + kt * 32, 448, lane);
      v16bf b = loadFragB(Btf1 + (size_t)nt * 16 * 448 + kt * 32, 448, lane);
      acc = wmma_bf16(a, b, acc);
    }
    int col = nt * 16 + (lane & 15);
    if (col < 100) {
      float bias = bf1[col];
      int rbase = (lane >> 4) * 8;
      for (int r = 0; r < 8; ++r) sfin[rbase + r][col] = acc[r] + bias;
    }
  }
  __syncthreads();
  // residual + LN (norm1) + vector residual/norm.  Half-wave (16 lanes) per node.
  {
    int nl = w * 2 + (lane >> 4);
    int n = nb + nl;
    int cl = lane & 15;
    float x[7], sum = 0.f, sumsq = 0.f;
    for (int k = 0; k < 7; ++k) {
      int col = k * 16 + cl;
      float v = 0.f;
      if (col < NSD) v = s_node[(size_t)n * NSD + col] + sfin[nl][col];
      x[k] = v; sum += v; sumsq += v * v;
    }
    for (int m = 8; m; m >>= 1) { sum += __shfl_xor(sum, m, 32); sumsq += __shfl_xor(sumsq, m, 32); }
    float mu = sum / NSD, var = sumsq / NSD - mu * mu;
    float rs = rsqrtf(var + 1e-5f);
    for (int k = 0; k < 7; ++k) {
      int col = k * 16 + cl;
      if (col < NSD) {
        float o = (x[k] - mu) * rs * gamma[col] + beta[col];
        s_node[(size_t)n * NSD + col] = o;
        s_bf[(size_t)n * NSP + col] = f2bf(o);
      }
    }
    float a0 = 0, a1 = 0, a2 = 0;
    for (int h = 0; h < 32; ++h) {
      float ww = wvf1[h * 16 + cl];
      a0 += ww * vhf[nl][h][0]; a1 += ww * vhf[nl][h][1]; a2 += ww * vhf[nl][h][2];
    }
    float* q = &v_node[(size_t)n * 48 + cl * 3];
    float w0 = q[0] + a0, w1 = q[1] + a1, w2 = q[2] + a2;
    float tot = fmaxf(w0 * w0 + w1 * w1 + w2 * w2, 1e-8f);
    for (int m = 8; m; m >>= 1) tot += __shfl_xor(tot, m, 32);
    float r = rsqrtf(tot * (1.f / 16.f));
    q[0] = w0 * r; q[1] = w1 * r; q[2] = w2 * r;
  }
}

// ---------------------------------------------------------------------------
// Output head: LN(wout_ln) + GVP(100,16 -> 100) + scatter-mean pooling
// ---------------------------------------------------------------------------
__global__ __launch_bounds__(256) void out_head(
    const float* __restrict__ s_node, const float* __restrict__ v_node,
    const int* __restrict__ gidx,
    const float* __restrict__ gamma, const float* __restrict__ beta,
    const float* __restrict__ whO, const float* __restrict__ wsO,
    const float* __restrict__ bsO,
    float* __restrict__ pooled, float* __restrict__ gcnt) {
  __shared__ float scat[8][116];
  __shared__ float vsh[8][16][3];
  int node = blockIdx.x * 8 + (threadIdx.x >> 5);   // grid sized exactly
  int lane = threadIdx.x & 31;
  int sl = threadIdx.x >> 5;
  float x[4], sum = 0.f, sumsq = 0.f;
  for (int k = 0; k < 4; ++k) {
    int col = lane + 32 * k;
    float v = (col < NSD) ? s_node[(size_t)node * NSD + col] : 0.f;
    x[k] = v; sum += v; sumsq += v * v;
  }
  for (int m = 16; m; m >>= 1) { sum += __shfl_xor(sum, m, 32); sumsq += __shfl_xor(sumsq, m, 32); }
  float mu = sum / NSD, var = sumsq / NSD - mu * mu;
  float rs = rsqrtf(var + 1e-5f);
  for (int k = 0; k < 4; ++k) {
    int col = lane + 32 * k;
    if (col < NSD) scat[sl][col] = (x[k] - mu) * rs * gamma[col] + beta[col];
  }
  float vv[3] = {0.f, 0.f, 0.f}, sq = 0.f;
  if (lane < 16) {
    for (int c = 0; c < 3; ++c) { vv[c] = v_node[(size_t)node * 48 + lane * 3 + c]; sq += vv[c] * vv[c]; }
    sq = fmaxf(sq, 1e-8f);
  }
  float tot = sq;
  for (int m = 8; m; m >>= 1) tot += __shfl_xor(tot, m, 32);
  if (lane < 16) {
    float rv = rsqrtf(tot * (1.f / 16.f));
    for (int c = 0; c < 3; ++c) vsh[sl][lane][c] = vv[c] * rv;
  }
  __syncthreads();
  if (lane < 16) {
    float a0 = 0, a1 = 0, a2 = 0;
    for (int i = 0; i < 16; ++i) {
      float ww = whO[i * 16 + lane];
      a0 += ww * vsh[sl][i][0]; a1 += ww * vsh[sl][i][1]; a2 += ww * vsh[sl][i][2];
    }
    scat[sl][100 + lane] = sqrtf(fmaxf(a0 * a0 + a1 * a1 + a2 * a2, 1e-8f));
  }
  __syncthreads();
  int g = gidx[node];
  for (int k = 0; k < 4; ++k) {
    int col = lane + 32 * k;
    if (col < NSD) {
      float a = bsO[col];
      for (int kk = 0; kk < 116; ++kk) a += scat[sl][kk] * wsO[kk * NSD + col];
      atomicAdd(&pooled[(size_t)g * NSD + col], a);
    }
  }
  if (lane == 0) atomicAdd(&gcnt[g], 1.f);
}

// --------------------------- tiny dense head -------------------------------
__global__ __launch_bounds__(256) void dense1(const float* __restrict__ pooled,
                                              const float* __restrict__ gcnt,
                                              const float* __restrict__ w1,
                                              const float* __restrict__ b1,
                                              float* __restrict__ h1) {
  int idx = blockIdx.x * 256 + threadIdx.x;
  if (idx >= NGRAPHS * 200) return;
  int g = idx / 200, o = idx % 200;
  float inv = 1.f / fmaxf(gcnt[g], 1.f);
  float a = b1[o];
  for (int k = 0; k < NSD; ++k) a += pooled[(size_t)g * NSD + k] * inv * w1[k * 200 + o];
  h1[idx] = fmaxf(a, 0.f);
}
__global__ __launch_bounds__(256) void dense2(const float* __restrict__ h1,
                                              const float* __restrict__ w2,
                                              const float* __restrict__ b2,
                                              float* __restrict__ h2) {
  int idx = blockIdx.x * 256 + threadIdx.x;
  if (idx >= NGRAPHS * 200) return;
  int g = idx / 200, o = idx % 200;
  float a = b2[o];
  for (int k = 0; k < 200; ++k) a += h1[g * 200 + k] * w2[k * 200 + o];
  h2[idx] = fmaxf(a, 0.f);
}
__global__ __launch_bounds__(256) void dense3(const float* __restrict__ h2,
                                              const float* __restrict__ w3,
                                              const float* __restrict__ b3,
                                              float* __restrict__ out) {
  int idx = blockIdx.x * 256 + threadIdx.x;
  if (idx >= NGRAPHS * 10) return;
  int g = idx / 10, o = idx % 10;
  float a = b3[o];
  for (int k = 0; k < 200; ++k) a += h2[g * 200 + k] * w3[k * 10 + o];
  out[idx] = a;
}

// ---------------------------------------------------------------------------
// Host driver.  Param leaf order assumes jax tree_flatten (sorted dict keys):
//  6..11  dense{b1,b2,b3,w1,w2,w3}
//  12+24l layer l: ff0{bs,wh,ws,wv} ff1{bs,wh,ws,wv} msg0..2{bs,wh,ws,wv}
//         norm0{beta,gamma} norm1{beta,gamma}
//  84..   we_gvp{bs,wh,ws,wv} we_ln{beta,gamma} wout_gvp{bs,wh,ws}
//         wout_ln{beta,gamma} wv_gvp{bs,wh,ws,wv} wv_ln{beta,gamma}
// ---------------------------------------------------------------------------
extern "C" void kernel_launch(void* const* d_in, const int* in_sizes, int n_in,
                              void* d_out, int out_size, void* d_ws, size_t ws_size,
                              hipStream_t stream) {
  (void)in_sizes; (void)n_in; (void)out_size; (void)ws_size;
  const float* node_s = (const float*)d_in[0];
  const float* node_v = (const float*)d_in[1];
  const float* edge_s = (const float*)d_in[2];
  const float* edge_v = (const float*)d_in[3];
  const int* eidx = (const int*)d_in[4];
  const int* gidx = (const int*)d_in[5];
  const int* src = eidx;
  const int* dst = eidx + NEDGES;
  auto P = [&](int i) -> const float* { return (const float*)d_in[i]; };

  char* base = (char*)d_ws;
  size_t off = 0;
  auto carve = [&](size_t bytes) -> void* {
    void* p = base + off;
    off = (off + bytes + 255) & ~(size_t)255;
    return p;
  };
  float* s_nd = (float*)carve((size_t)NNODES * NSD * 4);
  ushort_t* s_bf = (ushort_t*)carve((size_t)NNODES * NSP * 2);
  float* v_nd = (float*)carve((size_t)NNODES * 48 * 4);
  ushort_t* es_bf = (ushort_t*)carve((size_t)NEDGES * 32 * 2);
  float* ev_f = (float*)carve((size_t)NEDGES * 3 * 4);
  float* agg_s = (float*)carve((size_t)NNODES * NSD * 4);
  float* agg_v = (float*)carve((size_t)NNODES * 48 * 4);
  float* cntb = (float*)carve((size_t)NNODES * 4);
  float* pooled = (float*)carve((size_t)NGRAPHS * NSD * 4);
  float* gcnt = (float*)carve((size_t)NGRAPHS * 4);
  float* h1 = (float*)carve((size_t)NGRAPHS * 200 * 4);
  float* h2 = (float*)carve((size_t)NGRAPHS * 200 * 4);
  ushort_t *Bt0[3], *Bt1[3], *Bt2[3], *Btf0[3], *Btf1[3];
  for (int l = 0; l < 3; ++l) {
    Bt0[l] = (ushort_t*)carve(112 * 288 * 2);
    Bt1[l] = (ushort_t*)carve(112 * 128 * 2);
    Bt2[l] = (ushort_t*)carve(112 * 128 * 2);
    Btf0[l] = (ushort_t*)carve(416 * 160 * 2);
    Btf1[l] = (ushort_t*)carve(112 * 448 * 2);
  }

  // repack weights (transposed, padded, bf16)
  for (int l = 0; l < 3; ++l) {
    int LB = 12 + l * 24;
    pack_wT<<<(112 * 288 + 255) / 256, 256, 0, stream>>>(P(LB + 10), Bt0[l], 265, 100, 288, 112);
    pack_wT<<<(112 * 128 + 255) / 256, 256, 0, stream>>>(P(LB + 14), Bt1[l], 116, 100, 128, 112);
    pack_wT<<<(112 * 128 + 255) / 256, 256, 0, stream>>>(P(LB + 18), Bt2[l], 116, 100, 128, 112);
    pack_wT<<<(416 * 160 + 255) / 256, 256, 0, stream>>>(P(LB + 2), Btf0[l], 132, 400, 160, 416);
    pack_wT<<<(112 * 448 + 255) / 256, 256, 0, stream>>>(P(LB + 6), Btf1[l], 432, 100, 448, 112);
  }

  hipMemsetAsync(cntb, 0, (size_t)NNODES * 4, stream);
  count_edges<<<(NEDGES + 255) / 256, 256, 0, stream>>>(dst, cntb);
  node_embed<<<(NNODES + 255) / 256, 256, 0, stream>>>(
      node_s, node_v, P(100), P(99), P(96), P(97), P(95), P(98), s_nd, s_bf, v_nd);
  edge_embed<<<(NEDGES + 255) / 256, 256, 0, stream>>>(
      edge_s, edge_v, P(89), P(88), P(85), P(86), P(84), P(87), es_bf, ev_f);

  for (int l = 0; l < 3; ++l) {
    int LB = 12 + l * 24;
    hipMemsetAsync(agg_s, 0, (size_t)NNODES * NSD * 4, stream);
    hipMemsetAsync(agg_v, 0, (size_t)NNODES * 48 * 4, stream);
    edge_msg<<<NEDGES / 32, 256, 0, stream>>>(
        s_bf, v_nd, es_bf, ev_f, src, dst,
        P(LB + 9), P(LB + 8), P(LB + 11),
        P(LB + 13), P(LB + 12), P(LB + 15),
        P(LB + 17), P(LB + 16), P(LB + 19),
        Bt0[l], Bt1[l], Bt2[l], agg_s, agg_v);
    node_update<<<NNODES / 8, 256, 0, stream>>>(s_nd, s_bf, v_nd, agg_s, agg_v,
                                                cntb, P(LB + 21), P(LB + 20));
    node_ff<<<NNODES / 16, 256, 0, stream>>>(
        s_nd, s_bf, v_nd, P(LB + 1), P(LB + 0), P(LB + 3),
        P(LB + 5), P(LB + 4), P(LB + 7),
        Btf0[l], Btf1[l], P(LB + 23), P(LB + 22));
  }

  hipMemsetAsync(pooled, 0, (size_t)NGRAPHS * NSD * 4, stream);
  hipMemsetAsync(gcnt, 0, (size_t)NGRAPHS * 4, stream);
  out_head<<<NNODES / 8, 256, 0, stream>>>(s_nd, v_nd, gidx, P(94), P(93), P(91),
                                           P(92), P(90), pooled, gcnt);
  dense1<<<(NGRAPHS * 200 + 255) / 256, 256, 0, stream>>>(pooled, gcnt, P(9), P(6), h1);
  dense2<<<(NGRAPHS * 200 + 255) / 256, 256, 0, stream>>>(h1, P(10), P(7), h2);
  dense3<<<(NGRAPHS * 10 + 255) / 256, 256, 0, stream>>>(h2, P(11), P(8), (float*)d_out);
}
